// Layer_6270652252844
// MI455X (gfx1250) — matrix operand order repaired and checked by reading, not compiled, not run on hardware
//
#include <hip/hip_runtime.h>

#define ALPHA 0.2f
constexpr int B  = 1024;
constexpr int NF = 65536;   // full feature dim
constexpr int Nh = 32768;   // half (even/odd) dim
constexpr int H  = 128;     // hidden
constexpr int SPLITK = 16;  // K-chunks for GEMM1
constexpr int KCHUNK = Nh / SPLITK;     // 2048 halves per block
constexpr int KSTAGE = 128;             // halves staged to LDS per step
constexpr int NSTAGE = KCHUNK / KSTAGE; // 16
constexpr int LDSPAD = 8;               // +16B row pad -> conflict-free ds_load_b128

typedef __attribute__((ext_vector_type(16))) _Float16 v16h;
typedef __attribute__((ext_vector_type(8)))  _Float16 v8h;
typedef __attribute__((ext_vector_type(4)))  _Float16 v4h;
typedef __attribute__((ext_vector_type(8)))  float    v8f;

union V16H { v16h v; v8h h[2]; };

// ---- async global->LDS copy (CDNA5 ASYNCcnt path), guarded ----------------
#if __has_builtin(__builtin_amdgcn_global_load_async_to_lds_b128)
#define HAVE_ASYNC_LDS 1
typedef int v4i __attribute__((vector_size(16)));
typedef __attribute__((address_space(1))) v4i v4i_gbl;
typedef __attribute__((address_space(3))) v4i v4i_lds;
#endif

__device__ __forceinline__ void cp_b128(const _Float16* g, _Float16* l) {
#ifdef HAVE_ASYNC_LDS
  __builtin_amdgcn_global_load_async_to_lds_b128((v4i_gbl*)g, (v4i_lds*)l, 0, 0);
#else
  *(v8h*)l = *(const v8h*)g;   // global_load_b128 + ds_store_b128 fallback
#endif
}

template <int N>
__device__ __forceinline__ void wait_async() {
#ifdef HAVE_ASYNC_LDS
#if __has_builtin(__builtin_amdgcn_s_wait_asynccnt)
  __builtin_amdgcn_s_wait_asynccnt(N);
#else
  asm volatile("s_wait_asynccnt %0" :: "i"(N) : "memory");
#endif
#endif
}

// ---- WMMA fragment loaders (CDNA5 ISA 7.12.2 layouts, wave32) -------------
// A (16x32 f16, MxK): lane m = lane&15; k chunks [base2,base2+8) and
// [16+base2,16+base2+8), base2 = (lane>=16)*8.
__device__ __forceinline__ v16h load_a_frag(const _Float16* __restrict__ row,
                                            int k0, int lane) {
  const int base2 = (lane >> 4) << 3;
  V16H u;
  u.h[0] = *(const v8h*)(row + k0 + base2);
  u.h[1] = *(const v8h*)(row + k0 + 16 + base2);
  return u.v;
}

// A fragment from an LDS tile row (row = &tile[m][0], padded stride)
__device__ __forceinline__ v16h lds_a_frag(const _Float16* row, int k0, int base2) {
  V16H u;
  u.h[0] = *(const v8h*)(row + k0 + base2);        // ds_load_b128
  u.h[1] = *(const v8h*)(row + k0 + 16 + base2);   // ds_load_b128
  return u.v;
}

// B (32x16 f16, KxN): lane holds column n, rows k = [khalf,khalf+8) and [16+khalf,16+khalf+8).
// Weights stored transposed [n][k] -> two contiguous b128 loads.
__device__ __forceinline__ v16h load_b_frag(const _Float16* __restrict__ wt,
                                            size_t ldk, int n, int k0, int lane) {
  const int khalf = (lane >> 4) << 3;
  const _Float16* p = wt + (size_t)n * ldk + k0 + khalf;
  V16H u;
  u.h[0] = *(const v8h*)(p);
  u.h[1] = *(const v8h*)(p + 16);
  return u.v;
}

__device__ __forceinline__ v8f wmma16(v16h a, v16h b, v8f c) {
  return __builtin_amdgcn_wmma_f32_16x16x32_f16(false, a, false, b,
                                                (short)0, c, false, false);
}

// ---- prep: gather even columns of phi -> contiguous f16 -------------------
__global__ void k_gather_even(const float* __restrict__ phi,
                              _Float16* __restrict__ Aev) {
  int t  = blockIdx.x * blockDim.x + threadIdx.x;   // B * Nh/4 threads
  int r  = t >> 13;                                 // Nh/4 = 8192
  int j0 = (t & 8191) << 2;
  const float4* p = (const float4*)(phi + (size_t)r * NF + 2 * j0);
  float4 a = p[0], b = p[1];
  v4h o = { (_Float16)a.x, (_Float16)a.z, (_Float16)b.x, (_Float16)b.z };
  *(v4h*)(Aev + (size_t)r * Nh + j0) = o;
}

// ---- prep: W (KxN, f32, row-major) -> Wt (NxK, f16) -----------------------
__global__ void k_transpose_w(const float* __restrict__ W,
                              _Float16* __restrict__ Wt, int K, int Ncols) {
  long long idx = (long long)blockIdx.x * blockDim.x + threadIdx.x;
  if (idx >= (long long)K * Ncols) return;
  int n = (int)(idx / K);
  int k = (int)(idx % K);
  Wt[idx] = (_Float16)W[(size_t)k * Ncols + n];
}

// ---- GEMM1: H1 += Aev(1024x32768) @ W1(32768x128), both MLPs, split-K -----
__device__ __forceinline__ void atomic_store_tile(v8f acc, int job, int mtile,
                                                  int lane, float* __restrict__ Cs,
                                                  float* __restrict__ Ct) {
  float* C = (job & 1) ? Ct : Cs;
  const int col  = ((job >> 1) << 4) + (lane & 15);
  const int mofs = (lane >> 4) << 3;
#pragma unroll
  for (int v = 0; v < 8; ++v) {
    int row = mtile * 16 + v + mofs;
    unsafeAtomicAdd(&C[(size_t)row * H + col], acc[v]);
  }
}

__global__ void k_gemm1(const _Float16* __restrict__ Aev,
                        const _Float16* __restrict__ Ws,
                        const _Float16* __restrict__ Wt,
                        float* __restrict__ Cs, float* __restrict__ Ct) {
  __shared__ _Float16 atile[2][16][KSTAGE + LDSPAD];

  const int lane  = threadIdx.x & 31;
  const int w     = threadIdx.x >> 5;                 // 0..3
  const int mtile = blockIdx.x;                       // 0..63
  const int kbeg  = blockIdx.y * KCHUNK;
  const int mrow0 = mtile * 16;

  // cooperative copy mapping: 128 threads, 16B each, 8 rows x 16 segs / instr
  const int cr = threadIdx.x >> 4;          // 0..7
  const int cs = (threadIdx.x & 15) << 3;   // half-offset, 8 halves = 16B

  const int job0 = w, job1 = w + 4, job2 = w + 8, job3 = w + 12;
  const int n0 = ((job0 >> 1) << 4) + (lane & 15);
  const int n1 = ((job1 >> 1) << 4) + (lane & 15);
  const int n2 = ((job2 >> 1) << 4) + (lane & 15);
  const int n3 = ((job3 >> 1) << 4) + (lane & 15);
  const _Float16* w0 = (job0 & 1) ? Wt : Ws;
  const _Float16* w1 = (job1 & 1) ? Wt : Ws;
  const _Float16* w2 = (job2 & 1) ? Wt : Ws;
  const _Float16* w3 = (job3 & 1) ? Wt : Ws;

  v8f acc0 = {}, acc1 = {}, acc2 = {}, acc3 = {};

  // prologue: stage 0 into buffer 0
  cp_b128(Aev + (size_t)(mrow0 + cr) * Nh + kbeg + cs, &atile[0][cr][cs]);
  cp_b128(Aev + (size_t)(mrow0 + 8 + cr) * Nh + kbeg + cs, &atile[0][8 + cr][cs]);

  int buf = 0;
  for (int stage = 0; stage < NSTAGE; ++stage) {
    const int kbase = kbeg + stage * KSTAGE;
    if (stage + 1 < NSTAGE) {
      const int knext = kbase + KSTAGE;
      cp_b128(Aev + (size_t)(mrow0 + cr) * Nh + knext + cs, &atile[buf ^ 1][cr][cs]);
      cp_b128(Aev + (size_t)(mrow0 + 8 + cr) * Nh + knext + cs, &atile[buf ^ 1][8 + cr][cs]);
      wait_async<2>();          // oldest stage complete, next still in flight
    } else {
      wait_async<0>();
    }
    __syncthreads();

    const _Float16* arow = &atile[buf][lane & 15][0];
    const int base2 = (lane >> 4) << 3;

    // one-step software pipeline over the 4 k-steps of this stage
    v16h a_cur = lds_a_frag(arow, 0, base2);
    v16h b0 = load_b_frag(w0, Nh, n0, kbase, lane);
    v16h b1 = load_b_frag(w1, Nh, n1, kbase, lane);
    v16h b2 = load_b_frag(w2, Nh, n2, kbase, lane);
    v16h b3 = load_b_frag(w3, Nh, n3, kbase, lane);
#pragma unroll
    for (int kk = 0; kk < 4; ++kk) {
      v16h a_nxt, c0, c1, c2, c3;
      if (kk < 3) {
        const int kn = kbase + (kk + 1) * 32;
        a_nxt = lds_a_frag(arow, (kk + 1) * 32, base2);
        c0 = load_b_frag(w0, Nh, n0, kn, lane);
        c1 = load_b_frag(w1, Nh, n1, kn, lane);
        c2 = load_b_frag(w2, Nh, n2, kn, lane);
        c3 = load_b_frag(w3, Nh, n3, kn, lane);
      }
      acc0 = wmma16(a_cur, b0, acc0);
      acc1 = wmma16(a_cur, b1, acc1);
      acc2 = wmma16(a_cur, b2, acc2);
      acc3 = wmma16(a_cur, b3, acc3);
      if (kk < 3) { a_cur = a_nxt; b0 = c0; b1 = c1; b2 = c2; b3 = c3; }
    }
    __syncthreads();            // buffer `buf` free for reuse
    buf ^= 1;
  }

  atomic_store_tile(acc0, job0, mtile, lane, Cs, Ct);
  atomic_store_tile(acc1, job1, mtile, lane, Cs, Ct);
  atomic_store_tile(acc2, job2, mtile, lane, Cs, Ct);
  atomic_store_tile(acc3, job3, mtile, lane, Cs, Ct);
}

// ---- leaky-relu + f16 convert (layer-1 activations, s and t packed) -------
__global__ void k_leaky_cvt(const float* __restrict__ acc,
                            _Float16* __restrict__ out, int n) {
  int i = blockIdx.x * blockDim.x + threadIdx.x;
  if (i < n) {
    float x = acc[i];
    out[i] = (_Float16)(x > 0.f ? x : ALPHA * x);
  }
}

// ---- GEMM2: H2 = leaky(H1 @ W2), K=128, both MLPs fused -------------------
__global__ void k_gemm2(const _Float16* __restrict__ H1,      // [2][B][H]
                        const _Float16* __restrict__ W2s,
                        const _Float16* __restrict__ W2t,
                        _Float16* __restrict__ H2) {           // [2][B][H]
  const int lane = threadIdx.x & 31;
  const int w    = threadIdx.x >> 5;
  const int job  = blockIdx.x * 4 + w;        // 0..1023
  const int mlp   = job & 1;
  const int ntile = (job >> 1) & 7;
  const int mtile = job >> 4;

  const _Float16* arow = H1 + (size_t)mlp * B * H
                       + (size_t)(mtile * 16 + (lane & 15)) * H;
  const _Float16* wsel = mlp ? W2t : W2s;
  const int nb = ntile * 16 + (lane & 15);

  v8f acc = {};
#pragma unroll
  for (int kk = 0; kk < 4; ++kk) {
    v16h a = load_a_frag(arow, kk * 32, lane);
    v16h b = load_b_frag(wsel, H, nb, kk * 32, lane);
    acc = wmma16(a, b, acc);
  }
  _Float16* out = H2 + (size_t)mlp * B * H;
  const int col  = ntile * 16 + (lane & 15);
  const int mofs = (lane >> 4) << 3;
#pragma unroll
  for (int v = 0; v < 8; ++v) {
    float x = acc[v];
    x = x > 0.f ? x : ALPHA * x;
    out[(size_t)(mtile * 16 + v + mofs) * H + col] = (_Float16)x;
  }
}

// ---- layer 3 + coupling + interleaved scatter (fully fused) ---------------
__global__ void k_layer3_combine(const _Float16* __restrict__ H2,   // [2][B][H]
                                 const _Float16* __restrict__ W3s,  // [Nh][H]
                                 const _Float16* __restrict__ W3t,
                                 const float* __restrict__ bs3,
                                 const float* __restrict__ bt3,
                                 const float* __restrict__ phi,
                                 float* __restrict__ z) {
  const int lane  = threadIdx.x & 31;
  const int w     = threadIdx.x >> 5;        // 0..7
  const int mtile = blockIdx.y;              // 0..63
  const int ctile = blockIdx.x * 8 + w;      // 0..2047

  const _Float16* arows = H2 + (size_t)(mtile * 16 + (lane & 15)) * H;
  const _Float16* arowt = arows + (size_t)B * H;
  const int nb = ctile * 16 + (lane & 15);

  v8f accs = {}, acct = {};
#pragma unroll
  for (int kk = 0; kk < 4; ++kk) {
    v16h as = load_a_frag(arows, kk * 32, lane);
    accs = wmma16(as, load_b_frag(W3s, H, nb, kk * 32, lane), accs);
    v16h at = load_a_frag(arowt, kk * 32, lane);
    acct = wmma16(at, load_b_frag(W3t, H, nb, kk * 32, lane), acct);
  }

  const int j  = ctile * 16 + (lane & 15);
  const float sb = bs3[j];
  const float tb = bt3[j];
  const int mofs = (lane >> 4) << 3;
#pragma unroll
  for (int v = 0; v < 8; ++v) {
    const int row = mtile * 16 + v + mofs;
    const float2 pe = *(const float2*)(phi + (size_t)row * NF + 2 * j);
    const float s = accs[v] + sb;
    const float t = acct[v] + tb;
    float2 o;
    o.x = pe.x;                              // z_even = phi_even
    o.y = (pe.y - t) * __expf(-s);           // z_odd
    *(float2*)(z + (size_t)row * NF + 2 * j) = o;
  }
}

// ---------------------------------------------------------------------------
extern "C" void kernel_launch(void* const* d_in, const int* in_sizes, int n_in,
                              void* d_out, int out_size, void* d_ws, size_t ws_size,
                              hipStream_t stream) {
  const float* phi = (const float*)d_in[0];
  const float* Ws1 = (const float*)d_in[1];
  const float* Ws2 = (const float*)d_in[2];
  const float* Ws3 = (const float*)d_in[3];
  const float* bs3 = (const float*)d_in[4];
  const float* Wt1 = (const float*)d_in[5];
  const float* Wt2 = (const float*)d_in[6];
  const float* Wt3 = (const float*)d_in[7];
  const float* bt3 = (const float*)d_in[8];
  float* z = (float*)d_out;

  char* p = (char*)d_ws;
  auto take = [&](size_t bytes) {
    char* q = p;
    p += (bytes + 255) & ~(size_t)255;
    return q;
  };
  _Float16* Aev   = (_Float16*)take((size_t)B * Nh * 2);   // 64 MiB
  _Float16* W1s_t = (_Float16*)take((size_t)H * Nh * 2);   // 8 MiB
  _Float16* W1t_t = (_Float16*)take((size_t)H * Nh * 2);
  _Float16* W3s_t = (_Float16*)take((size_t)Nh * H * 2);
  _Float16* W3t_t = (_Float16*)take((size_t)Nh * H * 2);
  _Float16* W2s_t = (_Float16*)take((size_t)H * H * 2);
  _Float16* W2t_t = (_Float16*)take((size_t)H * H * 2);
  float*    H1acc = (float*)take((size_t)2 * B * H * 4);   // s then t
  _Float16* H1f16 = (_Float16*)take((size_t)2 * B * H * 2);
  _Float16* H2f16 = (_Float16*)take((size_t)2 * B * H * 2);

  (void)hipMemsetAsync(H1acc, 0, (size_t)2 * B * H * 4, stream);

  k_gather_even<<<(B * (Nh / 4)) / 256, 256, 0, stream>>>(phi, Aev);

  const int tW1 = (Nh * H + 255) / 256;
  const int tW2 = (H * H + 255) / 256;
  k_transpose_w<<<tW1, 256, 0, stream>>>(Ws1, W1s_t, Nh, H);
  k_transpose_w<<<tW1, 256, 0, stream>>>(Wt1, W1t_t, Nh, H);
  k_transpose_w<<<tW1, 256, 0, stream>>>(Ws3, W3s_t, H, Nh);
  k_transpose_w<<<tW1, 256, 0, stream>>>(Wt3, W3t_t, H, Nh);
  k_transpose_w<<<tW2, 256, 0, stream>>>(Ws2, W2s_t, H, H);
  k_transpose_w<<<tW2, 256, 0, stream>>>(Wt2, W2t_t, H, H);

  k_gemm1<<<dim3(B / 16, SPLITK), 128, 0, stream>>>(
      Aev, W1s_t, W1t_t, H1acc, H1acc + (size_t)B * H);

  k_leaky_cvt<<<(2 * B * H) / 256, 256, 0, stream>>>(H1acc, H1f16, 2 * B * H);

  k_gemm2<<<256, 128, 0, stream>>>(H1f16, W2s_t, W2t_t, H2f16);

  k_layer3_combine<<<dim3(Nh / 16 / 8, B / 16), 256, 0, stream>>>(
      H2f16, W3s_t, W3t_t, bs3, bt3, phi, z);
}
